// fretime_iTransformer_28948079575618
// MI455X (gfx1250) — matrix operand order repaired and checked by reading, not compiled, not run on hardware
//
#include <hip/hip_runtime.h>
#include <math.h>

// ---------------------------------------------------------------------------
// Types for CDNA5 WMMA (wave32): v_wmma_f32_16x16x32_f16
// ---------------------------------------------------------------------------
typedef _Float16 half_t;
typedef __attribute__((ext_vector_type(16))) _Float16 v16h;
typedef __attribute__((ext_vector_type(8)))  float    v8f;

union FragU { v16h v; unsigned int ui[8]; _Float16 h[16]; };
union H16U  { _Float16 h; unsigned short u; };

__device__ __forceinline__ v8f wmma_f16(v16h a, v16h b, v8f c) {
  // 8 args: (neg_a, A, neg_b, B, c_mod, C, reuse_a, reuse_b)
  return __builtin_amdgcn_wmma_f32_16x16x32_f16(false, a, false, b, (short)0, c,
                                                false, false);
}

// A-matrix fragment, 16x32 f16 (ISA 7.12.2): lane = M (mod 16), half = lane>>4
//   i<4 : K = 2i + 8*half ; i>=4 : K = 16 + 2(i-4) + 8*half
__device__ __forceinline__ v16h load_fragA(const half_t* __restrict__ A, long lda,
                                           long M, long row0, int kk0, int lane) {
  FragU f;
  long r = row0 + (lane & 15);
  int hl = lane >> 4;
  if (r < M) {
    const half_t* p = A + (size_t)r * lda + kk0 + 8 * hl;
    const unsigned int* q  = (const unsigned int*)p;
    const unsigned int* q2 = (const unsigned int*)(p + 16);
    f.ui[0] = q[0];  f.ui[1] = q[1];  f.ui[2] = q[2];  f.ui[3] = q[3];
    f.ui[4] = q2[0]; f.ui[5] = q2[1]; f.ui[6] = q2[2]; f.ui[7] = q2[3];
  } else {
#pragma unroll
    for (int i = 0; i < 8; ++i) f.ui[i] = 0u;
  }
  return f.v;
}

// Packed-B fragment: Bp[(kt*Npad + c)*32 + q], lane c = col0+(l&15),
// lane's 16 K-values: K = 8*half .. 8*half+7 and 16+8*half .. 16+8*half+7 -> 2x b128
__device__ __forceinline__ v16h load_fragBp(const half_t* __restrict__ Bp, long Npad,
                                            long kt, long col0, int lane) {
  long c = col0 + (lane & 15);
  if (c >= Npad) c = Npad - 1;  // discarded by epilogue guard
  const unsigned int* q =
      (const unsigned int*)(Bp + ((kt * Npad + c) << 5) + ((lane >> 4) << 3));
  FragU f;
#pragma unroll
  for (int i = 0; i < 8; ++i) f.ui[i] = q[i + ((i >> 2) << 2)];
  return f.v;
}

// ---------------------------------------------------------------------------
// Generic f16 WMMA GEMM: out = act(A(MxK) @ B(KxN) + bias + addsrc), K%32==0,
// N%16==0. block = 128 threads (4 waves); block tile 64x128; wave tile 32x64
// (2x4 fragments, 8 WMMA per K-step). B pre-packed fragment-contiguous.
// Optional outv: scatter f16 result into attention-packed V layout.
// ---------------------------------------------------------------------------
__global__ __launch_bounds__(128) void gemm_f16_kernel(
    const half_t* __restrict__ A, const half_t* __restrict__ Bp,
    const float* __restrict__ bias, const float* __restrict__ addsrc,
    float* __restrict__ outf, half_t* __restrict__ outh, half_t* __restrict__ outv,
    int M, int Ncol, int Kdim, int act, int accum) {
  int lane = threadIdx.x & 31;
  int wave = threadIdx.x >> 5;
  long m0 = (long)blockIdx.y * 64 + (wave >> 1) * 32;
  long n0 = (long)blockIdx.x * 128 + (wave & 1) * 64;

  v8f z = {0.f, 0.f, 0.f, 0.f, 0.f, 0.f, 0.f, 0.f};
  v8f acc[2][4] = {{z, z, z, z}, {z, z, z, z}};

  int KT = Kdim >> 5;
  for (int kt = 0; kt < KT; ++kt) {
    if (kt + 1 < KT) {  // L2-resident streams; emits global_prefetch_b8
      __builtin_prefetch(A + (size_t)(m0 + (lane & 15)) * Kdim + kt * 32 + 32, 0, 1);
      __builtin_prefetch(Bp + (((size_t)(kt + 1) * Ncol + n0 + (lane & 15)) << 5), 0, 1);
    }
    v16h a0 = load_fragA(A, Kdim, M, m0, kt * 32, lane);
    v16h a1 = load_fragA(A, Kdim, M, m0 + 16, kt * 32, lane);
    v16h b0 = load_fragBp(Bp, Ncol, kt, n0, lane);
    v16h b1 = load_fragBp(Bp, Ncol, kt, n0 + 16, lane);
    v16h b2 = load_fragBp(Bp, Ncol, kt, n0 + 32, lane);
    v16h b3 = load_fragBp(Bp, Ncol, kt, n0 + 48, lane);
    acc[0][0] = wmma_f16(a0, b0, acc[0][0]);
    acc[0][1] = wmma_f16(a0, b1, acc[0][1]);
    acc[0][2] = wmma_f16(a0, b2, acc[0][2]);
    acc[0][3] = wmma_f16(a0, b3, acc[0][3]);
    acc[1][0] = wmma_f16(a1, b0, acc[1][0]);
    acc[1][1] = wmma_f16(a1, b1, acc[1][1]);
    acc[1][2] = wmma_f16(a1, b2, acc[1][2]);
    acc[1][3] = wmma_f16(a1, b3, acc[1][3]);
  }

  int hl = lane >> 4, nl = lane & 15;
#pragma unroll
  for (int mi = 0; mi < 2; ++mi)
#pragma unroll
    for (int ni = 0; ni < 4; ++ni) {
      long coln = n0 + ni * 16 + nl;
      if (coln >= Ncol) continue;
      float bv = bias ? bias[coln] : 0.f;
#pragma unroll
      for (int j = 0; j < 8; ++j) {
        long row = m0 + mi * 16 + j + 8 * hl;
        if (row >= M) continue;
        long o = row * (long)Ncol + coln;
        float v = acc[mi][ni][j] + bv;
        if (addsrc) v += addsrc[o];
        if (act == 1) v = fmaxf(v, 0.f);
        else if (act == 2) v = 0.5f * v * (1.0f + erff(v * 0.70710678118f));
        if (outf) { if (accum) outf[o] += v; else outf[o] = v; }
        if (outh) outh[o] = (half_t)v;
        if (outv) {  // attention-packed V: (b,h, s-tile, e, s%32)
          int bb = (int)(row / 260), s = (int)(row % 260);
          long vo = ((((long)bb * 8 + (coln >> 6)) * 9 + (s >> 5)) * 64 +
                     (coln & 63)) * 32 + (s & 31);
          outv[vo] = (half_t)v;
        }
      }
    }
}

// ---------------------------------------------------------------------------
// Fused attention: per (b,h), 32 query rows/block. S=260, E=64, scale 1/8.
// Scores (32x288) staged in LDS (55KB << 320KB/WGP), softmax, P@V via WMMA.
// V consumed from fragment-packed layout (contiguous b128 loads).
// ---------------------------------------------------------------------------
__global__ __launch_bounds__(64) void attention_kernel(
    const half_t* __restrict__ Q, const half_t* __restrict__ Km,
    const half_t* __restrict__ Vp, half_t* __restrict__ O) {
  __shared__ alignas(16) float          sS[32][288];
  __shared__ alignas(16) unsigned short sP[32][288];  // f16 bit patterns

  int b = blockIdx.z, h = blockIdx.y;
  int row0 = blockIdx.x * 32;
  int lane = threadIdx.x & 31, wave = threadIdx.x >> 5;
  int hl = lane >> 4, nl = lane & 15;
  long base = ((long)b * 260) * 512 + h * 64;
  v8f z = {0.f, 0.f, 0.f, 0.f, 0.f, 0.f, 0.f, 0.f};

  // ---- Phase 1: S = (Q K^T) / 8 into LDS -------------------------------
  for (int nt = 0; nt < 17; ++nt) {
    v8f c = z;
    for (int ks = 0; ks < 64; ks += 32) {
      FragU fa;
      int r = row0 + wave * 16 + nl;
      if (r < 260) {
        const half_t* p = Q + base + (long)r * 512 + ks + 8 * hl;
        const unsigned int* q  = (const unsigned int*)p;
        const unsigned int* q2 = (const unsigned int*)(p + 16);
        fa.ui[0] = q[0];  fa.ui[1] = q[1];  fa.ui[2] = q[2];  fa.ui[3] = q[3];
        fa.ui[4] = q2[0]; fa.ui[5] = q2[1]; fa.ui[6] = q2[2]; fa.ui[7] = q2[3];
      } else {
#pragma unroll
        for (int i = 0; i < 8; ++i) fa.ui[i] = 0u;
      }
      FragU fb;
      int s = nt * 16 + nl;
      if (s < 260) {
        const half_t* p = Km + base + (long)s * 512 + ks + 8 * hl;
        const unsigned int* q = (const unsigned int*)p;
#pragma unroll
        for (int i = 0; i < 8; ++i) fb.ui[i] = q[i + ((i >> 2) << 2)];
      } else {
#pragma unroll
        for (int i = 0; i < 8; ++i) fb.ui[i] = 0u;
      }
      c = wmma_f16(fa.v, fb.v, c);
    }
#pragma unroll
    for (int j = 0; j < 8; ++j)
      sS[wave * 16 + j + 8 * hl][nt * 16 + nl] = c[j] * 0.125f;
  }
  __syncthreads();

  // ---- Phase 2: row softmax (cols >= 260 padded to 0 prob) --------------
  if (threadIdx.x < 32) {
    int r = threadIdx.x;
    float m = -1e30f;
    for (int cc = 0; cc < 260; ++cc) m = fmaxf(m, sS[r][cc]);
    float ssum = 0.f;
    for (int cc = 0; cc < 260; ++cc) { float e = expf(sS[r][cc] - m); sS[r][cc] = e; ssum += e; }
    float inv = 1.0f / ssum;
    for (int cc = 0; cc < 288; ++cc) {
      float pv = (cc < 260) ? sS[r][cc] * inv : 0.f;
      H16U u; u.h = (_Float16)pv; sP[r][cc] = u.u;
    }
  }
  __syncthreads();

  // ---- Phase 3: O = P @ V  (K padded 260 -> 288, Vp pads are zero) ------
  long vbase = (((long)b * 8 + h) * 9) * 64 * 32;
  for (int et = 0; et < 4; ++et) {
    v8f c = z;
    for (int kt = 0; kt < 9; ++kt) {
      FragU fa;
      {
        const unsigned short* p = &sP[wave * 16 + nl][kt * 32 + 8 * hl];
        const unsigned int* q  = (const unsigned int*)p;
        const unsigned int* q2 = (const unsigned int*)(p + 16);
        fa.ui[0] = q[0];  fa.ui[1] = q[1];  fa.ui[2] = q[2];  fa.ui[3] = q[3];
        fa.ui[4] = q2[0]; fa.ui[5] = q2[1]; fa.ui[6] = q2[2]; fa.ui[7] = q2[3];
      }
      FragU fb;
      {
        const unsigned int* q = (const unsigned int*)
            (Vp + vbase + ((long)kt * 64 + et * 16 + nl) * 32 + ((lane >> 4) << 3));
#pragma unroll
        for (int i = 0; i < 8; ++i) fb.ui[i] = q[i + ((i >> 2) << 2)];
      }
      c = wmma_f16(fa.v, fb.v, c);
    }
#pragma unroll
    for (int j = 0; j < 8; ++j) {
      int grow = row0 + wave * 16 + j + 8 * hl;
      if (grow < 260) O[base + (long)grow * 512 + et * 16 + nl] = (half_t)c[j];
    }
  }
}

// ---------------------------------------------------------------------------
// Elementwise / reduction kernels
// ---------------------------------------------------------------------------
__global__ void norm_kernel(const float* __restrict__ x, float* __restrict__ xc,
                            float* __restrict__ means, float* __restrict__ stdev) {
  int bn = blockIdx.x; int b = bn >> 7; int n = bn & 127;
  __shared__ float s1[128], s2[128];
  float a = 0.f, q = 0.f;
  for (int l = threadIdx.x; l < 512; l += 128) {
    float v = x[((long)b * 512 + l) * 128 + n];
    a += v; q += v * v;
  }
  s1[threadIdx.x] = a; s2[threadIdx.x] = q; __syncthreads();
  for (int o = 64; o > 0; o >>= 1) {
    if (threadIdx.x < o) { s1[threadIdx.x] += s1[threadIdx.x + o]; s2[threadIdx.x] += s2[threadIdx.x + o]; }
    __syncthreads();
  }
  float mean = s1[0] / 512.0f;
  float var  = s2[0] / 512.0f - mean * mean;
  float sd = sqrtf(fmaxf(var, 0.f) + 1e-5f);
  if (threadIdx.x == 0) { means[bn] = mean; stdev[bn] = sd; }
  float inv = 1.0f / sd;
  for (int l = threadIdx.x; l < 512; l += 128) {
    long o = ((long)b * 512 + l) * 128 + n;
    xc[o] = (x[o] - mean) * inv;
  }
}

__global__ void ln_kernel(const float* __restrict__ X, const float* __restrict__ g,
                          const float* __restrict__ bb, float* __restrict__ outf,
                          half_t* __restrict__ outh, int D) {
  int row = blockIdx.x;
  const float* x = X + (long)row * D;
  __shared__ float sred[256];
  float s = 0.f;
  for (int i = threadIdx.x; i < D; i += 256) s += x[i];
  sred[threadIdx.x] = s; __syncthreads();
  for (int o = 128; o > 0; o >>= 1) { if (threadIdx.x < o) sred[threadIdx.x] += sred[threadIdx.x + o]; __syncthreads(); }
  float mean = sred[0] / D; __syncthreads();
  float v = 0.f;
  for (int i = threadIdx.x; i < D; i += 256) { float d = x[i] - mean; v += d * d; }
  sred[threadIdx.x] = v; __syncthreads();
  for (int o = 128; o > 0; o >>= 1) { if (threadIdx.x < o) sred[threadIdx.x] += sred[threadIdx.x + o]; __syncthreads(); }
  float inv = rsqrtf(sred[0] / D + 1e-5f);
  for (int i = threadIdx.x; i < D; i += 256) {
    float y = (x[i] - mean) * inv * g[i] + bb[i];
    long o = (long)row * D + i;
    if (outf) outf[o] = y;
    if (outh) outh[o] = (half_t)y;
  }
}

__global__ void cast_f16_kernel(const float* __restrict__ s, half_t* __restrict__ d, long n) {
  long i = (long)blockIdx.x * blockDim.x + threadIdx.x;
  if (i < n) d[i] = (half_t)s[i];
}

__global__ void fill_h_kernel(half_t* __restrict__ d, long n) {
  long i = (long)blockIdx.x * blockDim.x + threadIdx.x;
  if (i < n) d[i] = (half_t)0.f;
}

// dense W (Kdim x Ncol, row-major) -> packed Bp[(kt*Ncol + c)*32 + q]
__global__ void pack_dense_kernel(const float* __restrict__ W, half_t* __restrict__ Bp,
                                  int Ncol, long total) {
  long idx = (long)blockIdx.x * blockDim.x + threadIdx.x;
  if (idx >= total) return;
  int qq = (int)(idx & 31);
  long t = idx >> 5;
  int c = (int)(t % Ncol);
  long kt = t / Ncol;
  Bp[idx] = (half_t)W[(kt * 32 + qq) * (long)Ncol + c];
}

// conv w (Cout, Cin, k) -> packed Bp with logical B[ci*k+j][co]
__global__ void pack_conv_kernel(const float* __restrict__ w, half_t* __restrict__ Bp,
                                 int Cout, int Cin, int k) {
  long total = (long)Cout * Cin * k;
  long idx = (long)blockIdx.x * blockDim.x + threadIdx.x;
  if (idx >= total) return;
  int qq = (int)(idx & 31);
  long t = idx >> 5;
  int c = (int)(t % Cout);
  long kt = t / Cout;
  int kk = (int)(kt * 32 + qq);
  int ci = kk / k, j = kk % k;
  Bp[idx] = (half_t)w[((long)c * Cin + ci) * k + j];
}

// causal im2col: src (32, Lt, C) -> dst[(b*Lt+t)*C*k + ci*k + j]
__global__ void im2col_kernel(const float* __restrict__ src, half_t* __restrict__ dst,
                              int Lt, int C, int k) {
  long total = (long)32 * Lt * C * k;
  long idx = (long)blockIdx.x * blockDim.x + threadIdx.x;
  if (idx >= total) return;
  int K = C * k;
  long m = idx / K; int r = (int)(idx % K);
  int ci = r / k, j = r % k;
  int b = (int)(m / Lt), t = (int)(m % Lt);
  int ts = t - (k - 1) + j;
  float v = (ts >= 0) ? src[((long)b * Lt + ts) * C + ci] : 0.f;
  dst[idx] = (half_t)v;
}

// STFT magnitudes: xc (32,512,128) -> mags[(b*1152 + c*9 + f)*129 + t]
__global__ void stft_kernel(const float* __restrict__ xc, float* __restrict__ mags) {
  long idx = (long)blockIdx.x * blockDim.x + threadIdx.x;
  if (idx >= (long)32 * 128 * 129) return;
  int t = (int)(idx % 129); int c = (int)((idx / 129) % 128); int b = (int)(idx / (129 * 128));
  float fr[16];
#pragma unroll
  for (int j = 0; j < 16; ++j) {
    int i = t * 4 + j - 8;
    if (i < 0) i = -i;
    if (i > 511) i = 1022 - i;
    float w = 0.5f - 0.5f * __cosf(6.28318530717958647f * (float)j / 16.0f);
    fr[j] = xc[((long)b * 512 + i) * 128 + c] * w;
  }
  long obase = ((long)b * 1152 + c * 9) * 129 + t;
  for (int f = 0; f < 9; ++f) {
    float re = 0.f, im = 0.f;
#pragma unroll
    for (int j = 0; j < 16; ++j) {
      float ang = -6.28318530717958647f * (float)(f * j) / 16.0f;
      float sn, cs; __sincosf(ang, &sn, &cs);
      re += fr[j] * cs; im += fr[j] * sn;
    }
    mags[obase + (long)f * 129] = sqrtf(re * re + im * im);
  }
}

// im2col for freq conv (valid, k=3): dst[(b*127+t)*3456 + cf*3 + j] = mags[b, cf, t+j]
__global__ void freq_im2col_kernel(const float* __restrict__ mags, half_t* __restrict__ dst) {
  long total = (long)32 * 127 * 3456;
  long idx = (long)blockIdx.x * blockDim.x + threadIdx.x;
  if (idx >= total) return;
  long m = idx / 3456; int r = (int)(idx % 3456);
  int cf = r / 3, j = r % 3;
  int b = (int)(m / 127), t = (int)(m % 127);
  dst[idx] = (half_t)mags[((long)b * 1152 + cf) * 129 + t + j];
}

// adaptive avg pool 127 -> 512 : y (32,127,128) -> xfre[(b*512+o)*128+c]
__global__ void pool_kernel(const float* __restrict__ y, float* __restrict__ xfre) {
  long idx = (long)blockIdx.x * blockDim.x + threadIdx.x;
  if (idx >= (long)32 * 512 * 128) return;
  int c = (int)(idx % 128); int o = (int)((idx / 128) % 512); int b = (int)(idx / (512 * 128));
  int start = (o * 127) >> 9;
  int end   = ((o + 1) * 127 + 511) >> 9;
  float s = 0.f;
  for (int t = start; t < end; ++t) s += y[((long)b * 127 + t) * 128 + c];
  xfre[((long)b * 512 + o) * 128 + c] = s / (float)(end - start);
}

// token matrix (f16 of (B,260,512)): ch<128 -> x_in, <256 -> x_fre, else marks
__global__ void tok_kernel(const float* __restrict__ xin, const float* __restrict__ xfre,
                           const float* __restrict__ xmark, half_t* __restrict__ tok) {
  long idx = (long)blockIdx.x * blockDim.x + threadIdx.x;
  if (idx >= (long)32 * 260 * 512) return;
  int t = (int)(idx % 512); int ch = (int)((idx / 512) % 260); int b = (int)(idx / (512 * 260));
  float v;
  if (ch < 128)      v = xin[((long)b * 512 + t) * 128 + ch];
  else if (ch < 256) v = xfre[((long)b * 512 + t) * 128 + (ch - 128)];
  else               v = xmark[((long)b * 512 + t) * 4 + (ch - 256)];
  tok[idx] = (half_t)v;
}

// out[b,p,n] = dec[(b*260+n)*96+p] * std[b,n] + mean[b,n]
__global__ void denorm_kernel(const float* __restrict__ dec, const float* __restrict__ sd,
                              const float* __restrict__ mn, float* __restrict__ out) {
  long idx = (long)blockIdx.x * blockDim.x + threadIdx.x;
  if (idx >= (long)32 * 96 * 128) return;
  int n = (int)(idx % 128); int p = (int)((idx / 128) % 96); int b = (int)(idx / (96 * 128));
  out[idx] = dec[((long)b * 260 + n) * 96 + p] * sd[b * 128 + n] + mn[b * 128 + n];
}

// ---------------------------------------------------------------------------
// Host orchestration
// ---------------------------------------------------------------------------
extern "C" void kernel_launch(void* const* d_in, const int* in_sizes, int n_in,
                              void* d_out, int out_size, void* d_ws, size_t ws_size,
                              hipStream_t stream) {
  if (n_in < 82) return;
  // pytree-flatten order (sorted dict keys):
  // 0 x_enc, 1 x_mark_enc, 2 W_emb, 3 W_proj, 4 b_emb, 5 b_proj, 6 bn,
  // 7 fre_b, 8 fre_w, 9 gn, 10.. layers (16 each: W1,W2,Wk,Wo,Wq,Wv,
  // b1,b2,bk,bo,bq,bv,c1,c2,g1,g2), 58.. tcn1/2/3 (8 each: b,br,w,wr x2)
  const float* x_enc  = (const float*)d_in[0];
  const float* x_mark = (const float*)d_in[1];
  const float* W_emb  = (const float*)d_in[2];
  const float* W_proj = (const float*)d_in[3];
  const float* b_emb  = (const float*)d_in[4];
  const float* b_proj = (const float*)d_in[5];
  const float* bn_g   = (const float*)d_in[6];
  const float* fre_b  = (const float*)d_in[7];
  const float* fre_w  = (const float*)d_in[8];
  const float* gn_g   = (const float*)d_in[9];
  auto LY = [&](int l, int w) -> const float* { return (const float*)d_in[10 + l * 16 + w]; };
  auto TC = [&](int br, int i) -> const float* { return (const float*)d_in[58 + br * 8 + i]; };

  // ---- workspace arena ---------------------------------------------------
  char* p = (char*)d_ws;
  auto alloc = [&](size_t bytes) -> char* {
    char* r = p; p += (bytes + 255) & ~(size_t)255; return r;
  };
  float*  means_f = (float*)alloc(4096 * 4);
  float*  stdev_f = (float*)alloc(4096 * 4);
  float*  xc_f    = (float*)alloc((size_t)2097152 * 4);     // (32,512,128)
  float*  xin_f   = (float*)alloc((size_t)2097152 * 4);
  float*  xfre_f  = (float*)alloc((size_t)2097152 * 4);
  half_t* xcbf    = (half_t*)alloc((size_t)2097152 * 2);
  float*  hf      = (float*)alloc((size_t)4194304 * 4);     // (32*512, 256)
  half_t* hbf     = (half_t*)alloc((size_t)4194304 * 2);
  float*  ybuf    = (float*)alloc((size_t)4194304 * 4);
  half_t* im2c    = (half_t*)alloc((size_t)29360128 * 2);   // 16384 x 1792 max
  half_t* wpack   = (half_t*)alloc((size_t)2097152 * 2);    // max 512x2048
  float*  mags_f  = (float*)alloc((size_t)4755456 * 4);     // (32,1152,129)
  float*  encf    = (float*)alloc((size_t)4259840 * 4);     // (8320,512)
  half_t* encbf   = (half_t*)alloc((size_t)4259840 * 2);
  half_t* qbf     = (half_t*)alloc((size_t)4259840 * 2);
  half_t* kbf     = (half_t*)alloc((size_t)4259840 * 2);
  half_t* vpack   = (half_t*)alloc((size_t)4718592 * 2);    // (32,8,9,64,32)
  half_t* obf     = (half_t*)alloc((size_t)4259840 * 2);
  float*  x1f     = (float*)alloc((size_t)4259840 * 4);
  float*  x1nf    = (float*)alloc((size_t)4259840 * 4);
  half_t* x1nbf   = (half_t*)alloc((size_t)4259840 * 2);
  float*  x2f     = (float*)alloc((size_t)4259840 * 4);
  float*  decf    = (float*)alloc((size_t)798720 * 4);      // (8320,96)
  half_t* tokbf   = im2c;                                   // alias (dead when used)
  half_t* midbf   = im2c;                                   // alias (8320x2048)

  auto gemm = [&](const half_t* Ab, const half_t* Bb, const float* bias,
                  const float* add, float* of, half_t* oh, half_t* ov,
                  int M, int Nc, int K, int act, int accum) {
    dim3 g((unsigned)((Nc + 127) / 128), (unsigned)((M + 63) / 64), 1);
    gemm_f16_kernel<<<g, 128, 0, stream>>>(Ab, Bb, bias, add, of, oh, ov, M, Nc, K, act, accum);
  };
  auto cast = [&](const float* s, half_t* d, long n) {
    cast_f16_kernel<<<(unsigned)((n + 255) / 256), 256, 0, stream>>>(s, d, n);
  };
  auto packd = [&](const float* w, half_t* d, int Kd, int Nc) {
    long n = (long)Kd * Nc;
    pack_dense_kernel<<<(unsigned)((n + 255) / 256), 256, 0, stream>>>(w, d, Nc, n);
  };
  auto packc = [&](const float* w, half_t* d, int Co, int Ci, int k) {
    long n = (long)Co * Ci * k;
    pack_conv_kernel<<<(unsigned)((n + 255) / 256), 256, 0, stream>>>(w, d, Co, Ci, k);
  };

  // ---- 1. normalize ------------------------------------------------------
  norm_kernel<<<4096, 128, 0, stream>>>(x_enc, xc_f, means_f, stdev_f);
  cast(xc_f, xcbf, 2097152);

  // ---- 2. TCN branches (im2col + WMMA GEMM), accumulate into xin_f -------
  const int ksz[3] = {3, 5, 7};
  for (int br = 0; br < 3; ++br) {
    int k = ksz[br];
    long n0 = (long)32 * 512 * 128 * k;
    im2col_kernel<<<(unsigned)((n0 + 255) / 256), 256, 0, stream>>>(xc_f, im2c, 512, 128, k);
    packc(TC(br, 2), wpack, 256, 128, k);                                   // w0
    gemm(im2c, wpack, TC(br, 0), nullptr, ybuf, nullptr, nullptr, 16384, 256, 128 * k, 1, 0);
    packc(TC(br, 3), wpack, 256, 128, 1);                                   // wr0
    gemm(xcbf, wpack, TC(br, 1), ybuf, hf, hbf, nullptr, 16384, 256, 128, 1, 0);
    long n1 = (long)32 * 512 * 256 * k;
    im2col_kernel<<<(unsigned)((n1 + 255) / 256), 256, 0, stream>>>(hf, im2c, 512, 256, k);
    packc(TC(br, 6), wpack, 128, 256, k);                                   // w1
    gemm(im2c, wpack, TC(br, 4), nullptr, ybuf, nullptr, nullptr, 16384, 128, 256 * k, 1, 0);
    packc(TC(br, 7), wpack, 128, 256, 1);                                   // wr1
    gemm(hbf, wpack, TC(br, 5), ybuf, xin_f, nullptr, nullptr, 16384, 128, 256, 1, br > 0 ? 1 : 0);
  }

  // ---- 3. frequency fusion ----------------------------------------------
  {
    long n = (long)32 * 128 * 129;
    stft_kernel<<<(unsigned)((n + 255) / 256), 256, 0, stream>>>(xc_f, mags_f);
    long n2 = (long)32 * 127 * 3456;
    freq_im2col_kernel<<<(unsigned)((n2 + 255) / 256), 256, 0, stream>>>(mags_f, im2c);
    packc(fre_w, wpack, 128, 1152, 3);
    gemm(im2c, wpack, fre_b, nullptr, ybuf, nullptr, nullptr, 4064, 128, 3456, 0, 0);
    long n3 = (long)32 * 512 * 128;
    pool_kernel<<<(unsigned)((n3 + 255) / 256), 256, 0, stream>>>(ybuf, xfre_f);
  }

  // ---- 4. tokens + embedding --------------------------------------------
  {
    long n = (long)32 * 260 * 512;
    tok_kernel<<<(unsigned)((n + 255) / 256), 256, 0, stream>>>(xin_f, xfre_f, x_mark, tokbf);
    packd(W_emb, wpack, 512, 512);
    gemm(tokbf, wpack, b_emb, nullptr, encf, encbf, nullptr, 8320, 512, 512, 0, 0);
  }

  // ---- 5. transformer encoder layers ------------------------------------
  fill_h_kernel<<<(unsigned)((4718592 + 255) / 256), 256, 0, stream>>>(vpack, 4718592);
  for (int l = 0; l < 3; ++l) {
    packd(LY(l, 4), wpack, 512, 512);  // Wq
    gemm(encbf, wpack, LY(l, 10), nullptr, nullptr, qbf, nullptr, 8320, 512, 512, 0, 0);
    packd(LY(l, 2), wpack, 512, 512);  // Wk
    gemm(encbf, wpack, LY(l, 8), nullptr, nullptr, kbf, nullptr, 8320, 512, 512, 0, 0);
    packd(LY(l, 5), wpack, 512, 512);  // Wv
    gemm(encbf, wpack, LY(l, 11), nullptr, nullptr, nullptr, vpack, 8320, 512, 512, 0, 0);
    attention_kernel<<<dim3(9, 8, 32), 64, 0, stream>>>(qbf, kbf, vpack, obf);
    packd(LY(l, 3), wpack, 512, 512);  // Wo
    gemm(obf, wpack, LY(l, 9), encf, x1f, nullptr, nullptr, 8320, 512, 512, 0, 0);
    ln_kernel<<<8320, 256, 0, stream>>>(x1f, LY(l, 14), LY(l, 6), x1nf, x1nbf, 512);
    packd(LY(l, 0), wpack, 512, 2048);  // W1
    gemm(x1nbf, wpack, LY(l, 12), nullptr, nullptr, midbf, nullptr, 8320, 2048, 512, 2, 0);
    packd(LY(l, 1), wpack, 2048, 512);  // W2
    gemm(midbf, wpack, LY(l, 13), x1nf, x2f, nullptr, nullptr, 8320, 512, 2048, 0, 0);
    ln_kernel<<<8320, 256, 0, stream>>>(x2f, LY(l, 15), LY(l, 7), encf, encbf, 512);
  }
  ln_kernel<<<8320, 256, 0, stream>>>(encf, gn_g, bn_g, x1nf, x1nbf, 512);

  // ---- 6. projection + denorm -------------------------------------------
  packd(W_proj, wpack, 512, 96);
  gemm(x1nbf, wpack, b_proj, nullptr, decf, nullptr, nullptr, 8320, 96, 512, 0, 0);
  {
    long n = (long)32 * 96 * 128;
    denorm_kernel<<<(unsigned)((n + 255) / 256), 256, 0, stream>>>(decf, stdev_f, means_f,
                                                                   (float*)d_out);
  }
  (void)in_sizes; (void)out_size; (void)ws_size;
}
